// madsormer_ae_13795434954998
// MI455X (gfx1250) — matrix operand order repaired and checked by reading, not compiled, not run on hardware
//
#include <hip/hip_runtime.h>

// ---------------------------------------------------------------------------
// madsormer on MI455X (gfx1250): flash-style NW Gaussian kernel attention.
// logits = bw^2 * (Q.K^T) - 0.5*bw^2*||k||^2  (||q||^2 cancels in softmax)
// Heavy math on v_wmma_f32_16x16x32_f16 (one 16-query tile per wave32).
// K/V blocks staged once per workgroup via async global->LDS copies
// (ASYNCcnt path) and shared by all 4 waves, cutting L2 traffic 4x.
// ---------------------------------------------------------------------------

typedef __attribute__((ext_vector_type(16))) _Float16 v16h;
typedef __attribute__((ext_vector_type(8)))  _Float16 v8h;
typedef __attribute__((ext_vector_type(8)))  float    v8f;
typedef int b128_t __attribute__((vector_size(16)));   // matches builtin param type

#define B_ 16
#define S_ 2048
#define D_ 4
#define H_ 256
#define F_ 512
#define EPS_ 1e-5f

union V16 { v16h v; v8h h[2]; };

// 16-byte global -> LDS transfer. Prefer the CDNA5 async-to-LDS instruction
// (tracked by ASYNCcnt, no VGPR round trip); fall back to a plain copy.
__device__ __forceinline__ void async_ld16(void* lds, const void* g) {
#if __has_builtin(__builtin_amdgcn_global_load_async_to_lds_b128)
  __builtin_amdgcn_global_load_async_to_lds_b128(
      (__attribute__((address_space(1))) b128_t*)(g),
      (__attribute__((address_space(3))) b128_t*)(lds), 0, 0);
#else
  *(v8h*)lds = *(const v8h*)g;
#endif
}

__device__ __forceinline__ void wait_async0() {
#if __has_builtin(__builtin_amdgcn_s_wait_asynccnt)
  __builtin_amdgcn_s_wait_asynccnt(0);
#else
  asm volatile("s_wait_asynccnt 0x0" ::: "memory");
#endif
}

// ---------------------------------------------------------------- concat ---
__global__ void concat_kernel(const float* __restrict__ KEY,
                              const float* __restrict__ VAL,
                              float* __restrict__ X) {
  int i = blockIdx.x * blockDim.x + threadIdx.x;
  if (i >= B_ * S_) return;
  X[i * 4 + 0] = KEY[i * 3 + 0];
  X[i * 4 + 1] = KEY[i * 3 + 1];
  X[i * 4 + 2] = KEY[i * 3 + 2];
  X[i * 4 + 3] = VAL[i];
}

// ------------------------------------------------------------ projections ---
// One block per (b,s); thread h computes q,k,v for that hidden channel.
// Emits Q,K in [B,S,H] f16 (contiguous along H for WMMA operand loads),
// V transposed to [B,H,S] f16 (contiguous along S for the attn@V B-operand),
// and the per-key softmax bias  -0.5*bw^2*||k||^2.
__global__ void proj_kernel(const float* __restrict__ X,
                            const float* __restrict__ Wq, const float* __restrict__ bq,
                            const float* __restrict__ Wk, const float* __restrict__ bk,
                            const float* __restrict__ Wv, const float* __restrict__ bv,
                            const float* __restrict__ bwp,
                            _Float16* __restrict__ Qh, _Float16* __restrict__ Kh,
                            _Float16* __restrict__ Vt, float* __restrict__ kbias) {
  __shared__ float part[8];
  const int bs = blockIdx.x;
  const int h  = threadIdx.x;
  const float* x = X + bs * 4;
  const float x0 = x[0], x1 = x[1], x2 = x[2], x3 = x[3];
  float q = bq[h] + x0 * Wq[h] + x1 * Wq[H_ + h] + x2 * Wq[2 * H_ + h] + x3 * Wq[3 * H_ + h];
  float k = bk[h] + x0 * Wk[h] + x1 * Wk[H_ + h] + x2 * Wk[2 * H_ + h] + x3 * Wk[3 * H_ + h];
  float v = bv[h] + x0 * Wv[h] + x1 * Wv[H_ + h] + x2 * Wv[2 * H_ + h] + x3 * Wv[3 * H_ + h];
  Qh[(size_t)bs * H_ + h] = (_Float16)q;
  Kh[(size_t)bs * H_ + h] = (_Float16)k;
  const int b = bs / S_, s = bs % S_;
  Vt[((size_t)b * H_ + h) * S_ + s] = (_Float16)v;
  // block-reduce ||k||^2 (wave32 butterfly + 8 partials)
  float kk = k * k;
  for (int off = 16; off; off >>= 1) kk += __shfl_xor(kk, off, 32);
  if ((threadIdx.x & 31) == 0) part[threadIdx.x >> 5] = kk;
  __syncthreads();
  if (threadIdx.x == 0) {
    float s2 = 0.f;
    for (int w = 0; w < 8; ++w) s2 += part[w];
    const float bw = *bwp;
    kbias[bs] = -0.5f * bw * bw * s2;
  }
}

// ---------------------------------------------------------- flash attention ---
// 4 waves per block, one 16-query tile per wave; all 4 tiles belong to the
// same batch (128 tiles/batch, 4/block), so per 32-key block the workgroup
// async-stages K (32x256 f16) and V^T (256x32 f16) into LDS once and every
// wave's WMMA B-operands come from ds_load instead of 4x redundant L2 reads.
__global__ void __launch_bounds__(128)
flash_kernel(const _Float16* __restrict__ Qh, const _Float16* __restrict__ Kh,
             const _Float16* __restrict__ Vt, const float* __restrict__ kbias,
             const float* __restrict__ bwp, float* __restrict__ AO) {
  __shared__ __align__(32) _Float16 Kl[32 * 256];   // [key][h]   16 KB
  __shared__ __align__(32) _Float16 Vl[256 * 32];   // [h][key]   16 KB
  __shared__ __align__(32) _Float16 Pt[4][16 * 32]; // per-wave P  4 KB
  const int lane = threadIdx.x & 31;
  const int wave = threadIdx.x >> 5;
  const int tile = blockIdx.x * 4 + wave;        // 0 .. B*(S/16)-1
  const int b    = tile >> 7;                    // uniform across the block
  const int qb   = (tile & 127) << 4;
  const int ln   = lane & 15;
  const int lh   = lane >> 4;                    // 0/1 half-wave
  const float bw  = *bwp;
  const float bw2 = bw * bw;

  // Q tile as 8 A-operands (16x32 f16 layout: lane M=ln; halves 0..15 at
  // K = lh*16 + h, per 32-wide H chunk).
  V16 qa[8];
  const _Float16* qp = Qh + ((size_t)b * S_ + qb + ln) * H_ + lh * 16;
  #pragma unroll
  for (int c = 0; c < 8; ++c) {
    qa[c].h[0] = *(const v8h*)(qp + c * 32);
    qa[c].h[1] = *(const v8h*)(qp + c * 32 + 8);
  }

  v8f acc[16];
  #pragma unroll
  for (int t = 0; t < 16; ++t) acc[t] = (v8f){};
  float mrow[8], lrow[8];
  #pragma unroll
  for (int r = 0; r < 8; ++r) { mrow[r] = -3.0e38f; lrow[r] = 0.f; }

  const _Float16* kbase = Kh + (size_t)b * S_ * H_;
  const _Float16* vbase = Vt + (size_t)b * H_ * S_;
  const float*    biasb = kbias + b * S_;
  const int tid = threadIdx.x;

  for (int kb = 0; kb < S_; kb += 32) {
    // ---- cooperative async stage of this key block into LDS ----
    __syncthreads();                       // everyone done reading previous block
    #pragma unroll
    for (int j = 0; j < 8; ++j) {
      const int id = j * 128 + tid;        // 0..1023, 16B chunks
      // K block: 32 rows x 512B  (row = id>>5, 16B chunk = id&31)
      async_ld16(Kl + (id >> 5) * H_ + (id & 31) * 8,
                 kbase + (size_t)(kb + (id >> 5)) * H_ + (id & 31) * 8);
      // V block: 256 rows x 64B  (row = id>>2, 16B chunk = id&3)
      async_ld16(Vl + (id >> 2) * 32 + (id & 3) * 8,
                 vbase + (size_t)(id >> 2) * S_ + kb + (id & 3) * 8);
    }
    wait_async0();
    __syncthreads();                       // staged tiles visible to all waves

    // ---- logits: two 16x16 tiles, inner dim H=256 in 8 chunks of 32 ----
    v8f s01[2];
    #pragma unroll
    for (int t = 0; t < 2; ++t) {
      v8f s = (v8f){};
      // B-operand (K^T 32x16): lane N = key ln, K-dim = lh*16 + 0..15 / chunk
      const _Float16* kp = Kl + (t * 16 + ln) * H_ + lh * 16;
      #pragma unroll
      for (int c = 0; c < 8; ++c) {
        const v16h kv = *(const v16h*)(kp + c * 32);
        s = __builtin_amdgcn_wmma_f32_16x16x32_f16(false, qa[c].v, false, kv,
                                                   (short)0, s, false, false);
      }
      s01[t] = s;
    }
    const float bias0 = biasb[kb + ln];
    const float bias1 = biasb[kb + 16 + ln];

    // ---- online softmax over this 32-key block ----
    float p0[8], p1[8];
    #pragma unroll
    for (int r = 0; r < 8; ++r) {
      const float l0 = bw2 * s01[0][r] + bias0;
      const float l1 = bw2 * s01[1][r] + bias1;
      float vmax = fmaxf(l0, l1);
      #pragma unroll
      for (int off = 1; off < 16; off <<= 1)
        vmax = fmaxf(vmax, __shfl_xor(vmax, off, 32));  // row = one 16-lane group
      const float mn = fmaxf(mrow[r], vmax);
      const float sc = __expf(mrow[r] - mn);
      const float e0 = __expf(l0 - mn);
      const float e1 = __expf(l1 - mn);
      float rs = e0 + e1;
      #pragma unroll
      for (int off = 1; off < 16; off <<= 1) rs += __shfl_xor(rs, off, 32);
      lrow[r] = lrow[r] * sc + rs;
      mrow[r] = mn;
      p0[r] = e0; p1[r] = e1;
      #pragma unroll
      for (int t = 0; t < 16; ++t) acc[t][r] *= sc;     // rescale running output
    }

    // ---- stage P (16x32 f16) through per-wave LDS to build A-layout ----
    _Float16* pw = &Pt[wave][0];
    #pragma unroll
    for (int r = 0; r < 8; ++r) {
      const int row = lh * 8 + r;                       // C-layout row
      pw[row * 32 + ln]      = (_Float16)p0[r];
      pw[row * 32 + 16 + ln] = (_Float16)p1[r];
    }
    // DS ops are in-order within a wave; safe wave-synchronous readback.
    V16 pa;
    pa.h[0] = *(const v8h*)(pw + ln * 32 + lh * 16);
    pa.h[1] = *(const v8h*)(pw + ln * 32 + lh * 16 + 8);

    // ---- P @ V : 16 output column tiles, K=32 keys, B-operand from LDS ----
    #pragma unroll
    for (int t = 0; t < 16; ++t) {
      const v16h vv = *(const v16h*)(Vl + (t * 16 + ln) * 32 + lh * 16);
      acc[t] = __builtin_amdgcn_wmma_f32_16x16x32_f16(false, pa.v, false, vv,
                                                      (short)0, acc[t], false, false);
    }
  }

  // ---- normalize and store attn_out [B,S,H] f32 ----
  #pragma unroll
  for (int r = 0; r < 8; ++r) {
    const float inv = 1.f / lrow[r];
    const int row = qb + lh * 8 + r;
    float* op = AO + ((size_t)b * S_ + row) * H_;
    #pragma unroll
    for (int t = 0; t < 16; ++t) op[t * 16 + ln] = acc[t][r] * inv;
  }
}

// -------------------------------------------------------------- epilogue ---
// attn_out@Wo + bo, residual, LN1, FFN(4->512->4, relu), residual, LN2.
__global__ void epilogue_kernel(const float* __restrict__ Xin, const float* __restrict__ AO,
                                const float* __restrict__ Wo, const float* __restrict__ bo,
                                const float* __restrict__ g1, const float* __restrict__ be1,
                                const float* __restrict__ W1, const float* __restrict__ b1,
                                const float* __restrict__ W2, const float* __restrict__ b2,
                                const float* __restrict__ g2, const float* __restrict__ be2,
                                float* __restrict__ Xout) {
  const int bs = blockIdx.x * blockDim.x + threadIdx.x;
  if (bs >= B_ * S_) return;
  const float* ao = AO + (size_t)bs * H_;
  float o0 = bo[0], o1 = bo[1], o2 = bo[2], o3 = bo[3];
  for (int h = 0; h < H_; ++h) {
    const float a = ao[h];
    const float4 w = ((const float4*)Wo)[h];
    o0 = fmaf(a, w.x, o0); o1 = fmaf(a, w.y, o1);
    o2 = fmaf(a, w.z, o2); o3 = fmaf(a, w.w, o3);
  }
  const float* x = Xin + bs * 4;
  float r0 = x[0] + o0, r1 = x[1] + o1, r2 = x[2] + o2, r3 = x[3] + o3;
  // LayerNorm 1
  float m = 0.25f * (r0 + r1 + r2 + r3);
  float d0 = r0 - m, d1 = r1 - m, d2 = r2 - m, d3 = r3 - m;
  float var = 0.25f * (d0 * d0 + d1 * d1 + d2 * d2 + d3 * d3);
  float inv = rsqrtf(var + EPS_);
  const float e0 = d0 * inv * g1[0] + be1[0], e1 = d1 * inv * g1[1] + be1[1];
  const float e2 = d2 * inv * g1[2] + be1[2], e3 = d3 * inv * g1[3] + be1[3];
  // FFN
  float y0 = b2[0], y1 = b2[1], y2 = b2[2], y3 = b2[3];
  for (int f = 0; f < F_; ++f) {
    float a = b1[f] + e0 * W1[f] + e1 * W1[F_ + f] + e2 * W1[2 * F_ + f] + e3 * W1[3 * F_ + f];
    a = fmaxf(a, 0.f);
    const float4 w = ((const float4*)W2)[f];
    y0 = fmaf(a, w.x, y0); y1 = fmaf(a, w.y, y1);
    y2 = fmaf(a, w.z, y2); y3 = fmaf(a, w.w, y3);
  }
  r0 = e0 + y0; r1 = e1 + y1; r2 = e2 + y2; r3 = e3 + y3;
  // LayerNorm 2
  m = 0.25f * (r0 + r1 + r2 + r3);
  d0 = r0 - m; d1 = r1 - m; d2 = r2 - m; d3 = r3 - m;
  var = 0.25f * (d0 * d0 + d1 * d1 + d2 * d2 + d3 * d3);
  inv = rsqrtf(var + EPS_);
  float* xo = Xout + bs * 4;
  xo[0] = d0 * inv * g2[0] + be2[0];
  xo[1] = d1 * inv * g2[1] + be2[1];
  xo[2] = d2 * inv * g2[2] + be2[2];
  xo[3] = d3 * inv * g2[3] + be2[3];
}

// --------------------------------------------------------------- final fc ---
__global__ void final_kernel(const float* __restrict__ X, const float* __restrict__ Wfc,
                             const float* __restrict__ bfc, float* __restrict__ out) {
  const int bs = blockIdx.x * blockDim.x + threadIdx.x;
  if (bs >= B_ * S_) return;
  const float* x = X + bs * 4;
  out[bs] = bfc[0] + x[0] * Wfc[0] + x[1] * Wfc[1] + x[2] * Wfc[2] + x[3] * Wfc[3];
}

// ------------------------------------------------------------------ launch ---
extern "C" void kernel_launch(void* const* d_in, const int* in_sizes, int n_in,
                              void* d_out, int out_size, void* d_ws, size_t ws_size,
                              hipStream_t stream) {
  (void)in_sizes; (void)n_in; (void)out_size; (void)ws_size;
  const float* KEY = (const float*)d_in[0];
  const float* VAL = (const float*)d_in[1];
  const float* Wq  = (const float*)d_in[2];
  const float* bq  = (const float*)d_in[3];
  const float* Wk  = (const float*)d_in[4];
  const float* bk  = (const float*)d_in[5];
  const float* Wv  = (const float*)d_in[6];
  const float* bv  = (const float*)d_in[7];
  const float* Wo  = (const float*)d_in[8];
  const float* bo  = (const float*)d_in[9];
  const float* bw  = (const float*)d_in[10];
  const float* g1  = (const float*)d_in[11];
  const float* be1 = (const float*)d_in[12];
  const float* W1  = (const float*)d_in[13];
  const float* b1  = (const float*)d_in[14];
  const float* W2  = (const float*)d_in[15];
  const float* b2  = (const float*)d_in[16];
  const float* g2  = (const float*)d_in[17];
  const float* be2 = (const float*)d_in[18];
  const float* Wfc = (const float*)d_in[19];
  const float* bfc = (const float*)d_in[20];

  char* ws = (char*)d_ws;
  size_t off = 0;
  auto carve = [&](size_t bytes) -> void* {
    void* p = ws + off;
    off = (off + bytes + 255) & ~(size_t)255;
    return p;
  };
  float*    X0    = (float*)carve((size_t)B_ * S_ * D_ * sizeof(float));
  float*    X1    = (float*)carve((size_t)B_ * S_ * D_ * sizeof(float));
  _Float16* Qh    = (_Float16*)carve((size_t)B_ * S_ * H_ * sizeof(_Float16));
  _Float16* Kh    = (_Float16*)carve((size_t)B_ * S_ * H_ * sizeof(_Float16));
  _Float16* Vt    = (_Float16*)carve((size_t)B_ * S_ * H_ * sizeof(_Float16));
  float*    kbias = (float*)carve((size_t)B_ * S_ * sizeof(float));
  float*    AO    = (float*)carve((size_t)B_ * S_ * H_ * sizeof(float));

  concat_kernel<<<(B_ * S_ + 255) / 256, 256, 0, stream>>>(KEY, VAL, X0);

  float* Xc = X0; float* Xn = X1;
  for (int i = 0; i < 2; ++i) {
    proj_kernel<<<B_ * S_, H_, 0, stream>>>(
        Xc, Wq + (size_t)i * D_ * H_, bq + (size_t)i * H_,
        Wk + (size_t)i * D_ * H_, bk + (size_t)i * H_,
        Wv + (size_t)i * D_ * H_, bv + (size_t)i * H_, bw + i,
        Qh, Kh, Vt, kbias);
    flash_kernel<<<(B_ * (S_ / 16)) / 4, 128, 0, stream>>>(Qh, Kh, Vt, kbias, bw + i, AO);
    epilogue_kernel<<<(B_ * S_ + 127) / 128, 128, 0, stream>>>(
        Xc, AO, Wo + (size_t)i * H_ * D_, bo + (size_t)i * D_,
        g1 + (size_t)i * D_, be1 + (size_t)i * D_,
        W1 + (size_t)i * D_ * F_, b1 + (size_t)i * F_,
        W2 + (size_t)i * F_ * D_, b2 + (size_t)i * D_,
        g2 + (size_t)i * D_, be2 + (size_t)i * D_, Xn);
    float* t = Xc; Xc = Xn; Xn = t;
  }
  final_kernel<<<(B_ * S_ + 127) / 128, 128, 0, stream>>>(Xc, Wfc, bfc, (float*)d_out);
}
